// AWX_55190329754261
// MI455X (gfx1250) — matrix operand order repaired and checked by reading, not compiled, not run on hardware
//
#include <hip/hip_runtime.h>

typedef _Float16     v16h __attribute__((ext_vector_type(16)));
typedef _Float16     v8h  __attribute__((ext_vector_type(8)));
typedef _Float16     v4h  __attribute__((ext_vector_type(4)));
typedef float        v8f  __attribute__((ext_vector_type(8)));
typedef int          v4i  __attribute__((ext_vector_type(4)));
typedef unsigned int v8u  __attribute__((ext_vector_type(8)));

#define AWX_B 64
#define AWX_L 4096
#define AWX_C 2048
#define AWX_EPS 1e-6f
#define LOG2E 1.4426950408889634f

#define SUBK 512                 // K sub-chunk staged in LDS
#define LDA  (SUBK + 8)          // +8 halves row pad -> conflict-free ds_load_b128
#define NIT  (SUBK / 32)         // 16 inner iterations (compile-time)

// ---------------------------------------------------------------------------
// Kernel 1: a[b,l] = (f16) sigmoid(x[b,l])^5   (512 KB f16 into workspace)
// ---------------------------------------------------------------------------
__global__ void awx_prep(const float* __restrict__ x, _Float16* __restrict__ a) {
    int t = blockIdx.x * blockDim.x + threadIdx.x;   // 65536 threads, 4 elems each
    float4 v = ((const float4*)x)[t];
    float f[4] = {v.x, v.y, v.z, v.w};
    v4h o;
#pragma unroll
    for (int i = 0; i < 4; ++i) {
        float e  = __builtin_amdgcn_exp2f(-LOG2E * f[i]);   // sigmoid
        float s  = __builtin_amdgcn_rcpf(1.0f + e);
        float s2 = s * s;
        o[i] = (_Float16)(s2 * s2 * s);                     // s^5
    }
    ((v4h*)a)[t] = o;
}

// ---------------------------------------------------------------------------
// Kernel 2: P[kh][b][c] = sum_{k in chunk} a[b,k]*R[c,k]  (raw f32 partials)
// - wave owns full M=64 strip of one 16-wide N tile for one K chunk
// - A sub-chunk (64 x 512 f16) staged in LDS, shared by the 8 waves (same kh):
//   A reads use DScnt (partial s_wait_dscnt), never drain the R stream
// - R stream software-pipelined 2 iterations deep; inner loop fully unrolled
//   (trip count 16) so SSA renaming removes buffer-rotation copies and each
//   load is consumed 2 iterations after issue
// - R {0,1} -> f16 {0,1.0}: pack pair + v_pk_mul_lo_u16 with 0x3C00
// ---------------------------------------------------------------------------
__global__ void awx_gemm(const _Float16* __restrict__ A,
                         const int* __restrict__ R,
                         float* __restrict__ P,
                         int ksplit, int kchunk) {
    __shared__ _Float16 lA[64 * LDA];                // ~65 KB

    const int tid  = threadIdx.x;
    const int lane = tid & 31;
    const int wave = tid >> 5;
    const int kh   = blockIdx.x % ksplit;            // K chunk id
    const int nt   = (blockIdx.x / ksplit) * 8 + wave; // N tile 0..127

    const int lrow = lane & 15;
    const int hi   = lane >> 4;

    // B 16-bit operand: lane n = lane%16 holds a contiguous 16-K chunk
    const int* rp = R + (size_t)(nt * 16 + lrow) * AWX_L + hi * 16;

    const int kbeg = kh * kchunk;
    const int kend = kbeg + kchunk;

    const unsigned int pk_one = 0x3C003C00u;         // packed f16 {1.0, 1.0}
    v8f acc[4] = {v8f{}, v8f{}, v8f{}, v8f{}};

    for (int kb = kbeg; kb < kend; kb += SUBK) {
        // Prime the R pipeline first: HBM loads in flight during A staging.
        v4i c0[4], c1[4];
#pragma unroll
        for (int j = 0; j < 4; ++j) c0[j] = *(const v4i*)(rp + kb + 4 * j);
#pragma unroll
        for (int j = 0; j < 4; ++j) c1[j] = *(const v4i*)(rp + kb + 32 + 4 * j);

        __syncthreads();                             // prior sub-chunk consumed
        // Stage A[0:64, kb:kb+SUBK) into LDS (16 b128 copies per thread).
#pragma unroll
        for (int j = 0; j < 64 * (SUBK / 8) / 256; ++j) {
            int i   = tid + j * 256;
            int row = i / (SUBK / 8);
            int cc  = (i % (SUBK / 8)) * 8;
            *(v4i*)(lA + row * LDA + cc) =
                *(const v4i*)(A + (size_t)row * AWX_L + kb + cc);
        }
        __syncthreads();

#pragma unroll
        for (int it = 0; it < NIT; ++it) {
            const int k = kb + it * 32;
            __builtin_prefetch(rp + k + 512, 0, 1);  // L2 warm, 16 iters ahead

            // Issue R loads two iterations ahead (folds away on the tail).
            const bool have_nx = (it + 2 < NIT);
            v4i nx[4];
            if (have_nx) {
#pragma unroll
                for (int j = 0; j < 4; ++j)
                    nx[j] = *(const v4i*)(rp + k + 64 + 4 * j);
            }

            // Convert current R fragment (loaded 2 iterations ago).
            int rv[16] = {c0[0][0], c0[0][1], c0[0][2], c0[0][3],
                          c0[1][0], c0[1][1], c0[1][2], c0[1][3],
                          c0[2][0], c0[2][1], c0[2][2], c0[2][3],
                          c0[3][0], c0[3][1], c0[3][2], c0[3][3]};
            v8u bits;
#pragma unroll
            for (int j = 0; j < 8; ++j) {
                unsigned int p = (unsigned int)rv[2 * j] |
                                 ((unsigned int)rv[2 * j + 1] << 16);
                unsigned int m;
                asm("v_pk_mul_lo_u16 %0, %1, %2" : "=v"(m) : "v"(p), "v"(pk_one));
                bits[j] = m;
            }
            v16h bf = __builtin_bit_cast(v16h, bits);

            // A operand from LDS: lanes 0-15 K {0-15}, 16-31 {16-31}
            const int kc = it * 32 + hi * 16;
#pragma unroll
            for (int mt = 0; mt < 4; ++mt) {         // one B fragment -> 4 WMMAs
                const _Float16* lp = lA + (size_t)(mt * 16 + lrow) * LDA + kc;
                v8h alo = *(const v8h*)(lp);
                v8h ahi = *(const v8h*)(lp + 8);
                v16h af;
#pragma unroll
                for (int i = 0; i < 8; ++i) { af[i] = alo[i]; af[8 + i] = ahi[i]; }
                acc[mt] = __builtin_amdgcn_wmma_f32_16x16x32_f16(
                    false, af, false, bf, (short)0, acc[mt], false, false);
            }

            // Pipeline advance: pure SSA renaming under full unroll (no movs).
#pragma unroll
            for (int j = 0; j < 4; ++j) c0[j] = c1[j];
            if (have_nx) {
#pragma unroll
                for (int j = 0; j < 4; ++j) c1[j] = nx[j];
            }
        }
    }

    // Store raw partials. D layout: lane n = lane%16, VGPR i -> M = i+8*hi.
    float* Pk = P + (size_t)kh * (AWX_B * AWX_C);
    const int n = nt * 16 + lrow;
#pragma unroll
    for (int mt = 0; mt < 4; ++mt) {
        const int m0 = mt * 16 + hi * 8;
#pragma unroll
        for (int i = 0; i < 8; ++i)
            Pk[(size_t)(m0 + i) * AWX_C + n] = acc[mt][i];
    }
}

// ---------------------------------------------------------------------------
// Kernel 3: out = clamp(sum_kh P[kh], eps, 1-eps)^(1/5)
// ---------------------------------------------------------------------------
__global__ void awx_finalize(const float* __restrict__ P,
                             float* __restrict__ out, int ksplit) {
    int t = blockIdx.x * blockDim.x + threadIdx.x;   // 32768 threads, 4 elems
    float4 s = ((const float4*)P)[t];
    for (int kh = 1; kh < ksplit; ++kh) {
        float4 q = ((const float4*)P)[t + (size_t)kh * (AWX_B * AWX_C / 4)];
        s.x += q.x; s.y += q.y; s.z += q.z; s.w += q.w;
    }
    float v[4] = {s.x, s.y, s.z, s.w};
    float o[4];
#pragma unroll
    for (int i = 0; i < 4; ++i) {
        float c = fminf(fmaxf(v[i], AWX_EPS), 1.0f - AWX_EPS);
        o[i] = __builtin_amdgcn_exp2f(0.2f * __builtin_amdgcn_logf(c));
    }
    ((float4*)out)[t] = make_float4(o[0], o[1], o[2], o[3]);
}

// ---------------------------------------------------------------------------
// Launch. d_in[0] = inputs f32 [64,4096], d_in[1] = R int32 [2048,4096],
// d_out = f32 [64,2048]. Workspace: 512 KB A + ksplit*512 KB partials,
// ksplit chosen (deterministically) from ws_size.
// ---------------------------------------------------------------------------
extern "C" void kernel_launch(void* const* d_in, const int* in_sizes, int n_in,
                              void* d_out, int out_size, void* d_ws, size_t ws_size,
                              hipStream_t stream) {
    const float* x  = (const float*)d_in[0];
    const int*   R  = (const int*)d_in[1];
    float*       o  = (float*)d_out;

    const size_t ABYTES = (size_t)AWX_B * AWX_L * sizeof(_Float16);  // 512 KB
    const size_t PBYTES = (size_t)AWX_B * AWX_C * sizeof(float);     // 512 KB

    int ksplit = 1;                                   // K-split for occupancy
    if      (ws_size >= ABYTES + 8 * PBYTES) ksplit = 8;
    else if (ws_size >= ABYTES + 4 * PBYTES) ksplit = 4;
    else if (ws_size >= ABYTES + 2 * PBYTES) ksplit = 2;

    _Float16* a = (_Float16*)d_ws;
    float*    P = (float*)((char*)d_ws + ABYTES);

    awx_prep<<<(AWX_B * AWX_L / 4) / 256, 256, 0, stream>>>(x, a);
    awx_gemm<<<16 * ksplit, 256, 0, stream>>>(a, R, P, ksplit, AWX_L / ksplit);
    awx_finalize<<<(AWX_B * AWX_C / 4) / 256, 256, 0, stream>>>(P, o, ksplit);
}
